// LNDP_89275190214788
// MI455X (gfx1250) — matrix operand order repaired and checked by reading, not compiled, not run on hardware
//
#include <hip/hip_runtime.h>
#include <hip/hip_bf16.h>
#include <math.h>

// ---- problem constants (from reference) ----
#define NN   1536
#define OBSD 17
#define ACTD 6
// NF=8, EF=4, T=5, HEADS=3, QK=4, VF=8, EIN=7, GIN=19

typedef _Float16 f16;
typedef _Float16 v16h __attribute__((ext_vector_type(16)));
typedef _Float16 v8h  __attribute__((ext_vector_type(8)));
typedef float    v8f  __attribute__((ext_vector_type(8)));

__device__ __forceinline__ float sigm(float x) { return 1.0f / (1.0f + __expf(-x)); }

// ---------------- degree kernels ----------------
__global__ void k_rowsum(const float* __restrict__ A, float* __restrict__ isum) {
    __shared__ float red[256];
    int row = blockIdx.x;
    float acc = 0.f;
    for (int j = threadIdx.x; j < NN; j += 256) acc += A[row * NN + j];
    red[threadIdx.x] = acc; __syncthreads();
    for (int s = 128; s > 0; s >>= 1) {
        if (threadIdx.x < s) red[threadIdx.x] += red[threadIdx.x + s];
        __syncthreads();
    }
    if (threadIdx.x == 0) isum[row] = red[0];
}

__global__ void k_colsum(const float* __restrict__ A, float* __restrict__ osum) {
    int col = blockIdx.x * 256 + threadIdx.x;  // coalesced down columns
    float acc = 0.f;
    for (int i = 0; i < NN; ++i) acc += A[i * NN + col];
    osum[col] = acc;
}

// ---------------- node prep: x = Wpre@g, then q,k and V in WMMA-B layout ----------------
// V fragment image: vb[((h*48 + tile)*32 + lane)*16 + e], lane = col + ((p>>3)&1 ? 16 : 0),
// e = (p&7) + 8*(p>>4), p = node%32; cols 8..15 zero-padded so the attention B-load is branch-free.
__global__ void k_node_prep(const float* __restrict__ isum, const float* __restrict__ osum,
                            const float* __restrict__ aseq0, const float* __restrict__ h0,
                            const float* __restrict__ Wpre_w, const float* __restrict__ Wpre_b,
                            const float* __restrict__ Wq, const float* __restrict__ Wk,
                            const float* __restrict__ Wv,
                            float* __restrict__ qb, float* __restrict__ kb, f16* __restrict__ vb) {
    int n = blockIdx.x * 256 + threadIdx.x;
    if (n >= NN) return;
    float g[19];
    float idv = isum[n] * 0.1f, odv = osum[n] * 0.1f;
    g[0] = idv; g[1] = odv; g[2] = idv + odv * 0.05f;          // d = i_d + o_d/20
    g[3] = (n < OBSD) ? 1.f : 0.f;
    g[4] = (n >= NN - ACTD) ? 1.f : 0.f;
    for (int t = 0; t < 6; ++t) g[5 + t] = aseq0[n * 6 + t];
    for (int f = 0; f < 8; ++f) g[11 + f] = h0[n * 8 + f];
    float x[8];
    for (int f = 0; f < 8; ++f) {
        float a = Wpre_b[f];
        for (int c = 0; c < 19; ++c) a += Wpre_w[f * 19 + c] * g[c];
        x[f] = a;
    }
    int tile = n >> 5, p = n & 31;
    int hi = ((p >> 3) & 1) ? 16 : 0;   // lane group for this K row
    int ep = (p & 7) + ((p >> 4) << 3); // element slot within lane
    for (int h = 0; h < 3; ++h) {
        for (int qq = 0; qq < 4; ++qq) {
            float aq = 0.f, ak = 0.f;
            for (int f = 0; f < 8; ++f) {
                aq += Wq[h * 32 + qq * 8 + f] * x[f];
                ak += Wk[h * 32 + qq * 8 + f] * x[f];
            }
            qb[n * 12 + h * 4 + qq] = aq;
            kb[n * 12 + h * 4 + qq] = ak;
        }
        f16* base = vb + ((size_t)(h * 48 + tile) * 32) * 16;
        for (int c = 0; c < 8; ++c) {
            float av = 0.f;
            for (int f = 0; f < 8; ++f) av += Wv[h * 64 + c * 8 + f] * x[f];
            base[(c + hi) * 16 + ep]     = (f16)av;    // real column c
            base[(c + 8 + hi) * 16 + ep] = (f16)0.0f;  // zero pad column c+8
        }
    }
}

// ---------------- attention: scores + softmax + WMMA att@v ----------------
// grid = (NN/16, HEADS), block = 256 (8 waves). One 16-row tile × one head per WG.
__global__ void __launch_bounds__(256) k_attn(const float* __restrict__ qb,
                                              const float* __restrict__ kb,
                                              const f16* __restrict__ vb,
                                              const float* __restrict__ e0,
                                              const float* __restrict__ A,
                                              const float* __restrict__ We,
                                              float* __restrict__ xt) {
    __shared__ f16  sc[16 * 1536];   // scores, then exp(score-rowmax), f16 (48 KB)
    __shared__ float red[256];
    __shared__ float rowmax[16];
    __shared__ float rowsum[16];
    __shared__ float outacc[256];    // 16x16 output tile accumulator

    int tid = threadIdx.x;
    int i0 = blockIdx.x * 16;
    int h  = blockIdx.y;
    int i_loc = tid >> 4, jl = tid & 15;
    int i = i0 + i_loc;

    float q0 = qb[i * 12 + h * 4 + 0], q1 = qb[i * 12 + h * 4 + 1];
    float q2 = qb[i * 12 + h * 4 + 2], q3 = qb[i * 12 + h * 4 + 3];
    float w0 = We[h * 7 + 0], w1 = We[h * 7 + 1], w2 = We[h * 7 + 2], w3 = We[h * 7 + 3];
    float w4 = We[h * 7 + 4], w5 = We[h * 7 + 5], w6 = We[h * 7 + 6];

    // pass 1: scores (one streaming pass over e0 row-tile), track row max
    float lmax = -1e30f;
    for (int s = 0; s < 96; ++s) {
        int j = s * 16 + jl;
        float4 e4 = reinterpret_cast<const float4*>(e0)[i * NN + j];
        float k0 = kb[j * 12 + h * 4 + 0], k1 = kb[j * 12 + h * 4 + 1];
        float k2 = kb[j * 12 + h * 4 + 2], k3 = kb[j * 12 + h * 4 + 3];
        float sv = 0.5f * (q0 * k0 + q1 * k1 + q2 * k2 + q3 * k3)
                 + w0 * e4.x + w1 * e4.y + w2 * e4.z + w3 * e4.w
                 + w4 * A[i * NN + j] + w5 * A[j * NN + i]
                 + ((i == j) ? w6 : 0.f);
        sc[i_loc * 1536 + j] = (f16)sv;
        lmax = fmaxf(lmax, sv);
    }
    red[tid] = lmax;
    __syncthreads();
    if (tid < 16) {
        float m = -1e30f;
        for (int t = 0; t < 16; ++t) m = fmaxf(m, red[tid * 16 + t]);
        rowmax[tid] = m;
    }
    __syncthreads();

    // pass 2: exponentiate in place (exp paid once) + softmax denominator.
    // Denominator accumulates the f16-rounded values so it matches the WMMA
    // numerator exactly; each (i_loc, j) cell is owned by exactly one thread.
    float mrow = rowmax[i_loc];
    float lsum = 0.f;
    for (int s = 0; s < 96; ++s) {
        int j = s * 16 + jl;
        f16 ex = (f16)__expf((float)sc[i_loc * 1536 + j] - mrow);
        sc[i_loc * 1536 + j] = ex;
        lsum += (float)ex;
    }
    red[tid] = lsum;
    __syncthreads();
    if (tid < 16) {
        float ss = 0.f;
        for (int t = 0; t < 16; ++t) ss += red[tid * 16 + t];
        rowsum[tid] = ss;
    }
    outacc[tid] = 0.f;
    __syncthreads();

    // pass 3: out(16x16) = att_exp x Vpad, split-K across 8 waves.
    // Pure ds_load_b128 + global_load_b128 + v_wmma: 6 fully-unrolled
    // v_wmma_f32_16x16x32_f16 per wave, dual accumulators.
    int wave = tid >> 5, lane = tid & 31;
    int aM  = lane & 15;                 // A layout: lanes 0-15 / 16-31 both carry M=lane%16
    int kb1 = (lane < 16) ? 0 : 8;       // low-lane K base per ISA 16-bit 16x32 A table
    const f16* scrow = &sc[aM * 1536];
    v8f c0 = {}, c1 = {};
#pragma unroll
    for (int u = 0; u < 6; ++u) {        // uniform: EXEC stays all-ones for WMMA
        int s = wave + u * 8;
        int j0 = s * 32;
        v8h alo = *reinterpret_cast<const v8h*>(scrow + j0 + kb1);       // K = kb1+0..7
        v8h ahi = *reinterpret_cast<const v8h*>(scrow + j0 + 16 + kb1);  // K = 16+kb1+0..7
        v16h af = __builtin_shufflevector(alo, ahi, 0, 1, 2, 3, 4, 5, 6, 7,
                                          8, 9, 10, 11, 12, 13, 14, 15);
        v16h bf = *reinterpret_cast<const v16h*>(vb + ((size_t)(h * 48 + s) * 32 + lane) * 16);
        if (u & 1) c1 = __builtin_amdgcn_wmma_f32_16x16x32_f16(false, af, false, bf,
                                                               (short)0, c1, false, false);
        else       c0 = __builtin_amdgcn_wmma_f32_16x16x32_f16(false, af, false, bf,
                                                               (short)0, c0, false, false);
    }
    v8f c = c0 + c1;
#pragma unroll
    for (int rI = 0; rI < 8; ++rI) {     // C layout: M = rI + 8*(lane/16), N = lane%16
        int m = rI + ((lane >> 4) << 3);
        int n = lane & 15;
        atomicAdd(&outacc[m * 16 + n], c[rI]);  // ds_add_f32 split-K combine
    }
    __syncthreads();
    if (tid < 128) {                      // normalize by softmax denom, concat(head) layout
        int m = tid >> 3, n2 = tid & 7;
        xt[(i0 + m) * 24 + h * 8 + n2] = outacc[m * 16 + n2] / rowsum[m];
    }
}

// ---------------- node GRU + separable edge-GRU input vectors ----------------
__global__ void k_node_gru(const float* __restrict__ xt, const float* __restrict__ h0,
                           const float* __restrict__ Wo_w, const float* __restrict__ Wo_b,
                           const float* __restrict__ nwih, const float* __restrict__ nwhh,
                           const float* __restrict__ nbih, const float* __restrict__ nbhh,
                           const float* __restrict__ ewih, const float* __restrict__ ebih,
                           const float* __restrict__ aseq0, const float* __restrict__ rr,
                           float* __restrict__ gij, float* __restrict__ gii) {
    int n = blockIdx.x * 256 + threadIdx.x;
    if (n >= NN) return;
    float xo[8];
    for (int f = 0; f < 8; ++f) {
        float a = Wo_b[f];
        for (int c = 0; c < 24; ++c) a += Wo_w[f * 24 + c] * xt[n * 24 + c];
        xo[f] = a;
    }
    float gi[24], gh[24];
    for (int g = 0; g < 24; ++g) {
        float a = nbih[g], b = nbhh[g];
        for (int f = 0; f < 8; ++f) {
            a += nwih[g * 8 + f] * xo[f];
            b += nwhh[g * 8 + f] * h0[n * 8 + f];
        }
        gi[g] = a; gh[g] = b;
    }
    float h[8];
    for (int g = 0; g < 8; ++g) {
        float rg  = sigm(gi[g] + gh[g]);
        float z   = sigm(gi[8 + g] + gh[8 + g]);
        float nn2 = tanhf(gi[16 + g] + rg * gh[16 + g]);
        h[g] = (1.f - z) * nn2 + z * h0[n * 8 + g];
    }
    // gi[i,j] = u[j] + v[i]  (bias + r-term folded into v)
    float r0 = rr[0];
    for (int g = 0; g < 12; ++g) {
        float aj = 0.f;
        float ai = ebih[g] + ewih[g * 29 + 28] * r0;
        for (int f = 0; f < 8; ++f) {
            aj += ewih[g * 29 + f] * h[f];
            ai += ewih[g * 29 + 8 + f] * h[f];
        }
        for (int t = 0; t < 6; ++t) {
            aj += ewih[g * 29 + 16 + t] * aseq0[n * 6 + t];
            ai += ewih[g * 29 + 22 + t] * aseq0[n * 6 + t];
        }
        gij[n * 12 + g] = aj;
        gii[n * 12 + g] = ai;
    }
}

// ---------------- per-edge GRU (bandwidth bound: e0 in, e out) ----------------
__global__ void k_edge(const float* __restrict__ e0, const float* __restrict__ A,
                       const float* __restrict__ gij, const float* __restrict__ gii,
                       const float* __restrict__ ewhh, const float* __restrict__ ebhh,
                       float* __restrict__ eout, float* __restrict__ wbuf) {
    int i = blockIdx.x / 6;                       // NN/256 = 6 blocks per row -> gii[i] uniform
    int j = (blockIdx.x % 6) * 256 + threadIdx.x;
    size_t eid = (size_t)i * NN + j;
    float4 e4 = reinterpret_cast<const float4*>(e0)[eid];
    float e0v[4] = { e4.x, e4.y, e4.z, e4.w };
    float gh[12];
    for (int g = 0; g < 12; ++g) {
        float a = ebhh[g];
        for (int f = 0; f < 4; ++f) a += ewhh[g * 4 + f] * e0v[f];
        gh[g] = a;
    }
    float aij = A[eid];
    float out[4];
    for (int g = 0; g < 4; ++g) {
        float gi0 = gij[j * 12 + g]     + gii[i * 12 + g];
        float gi1 = gij[j * 12 + 4 + g] + gii[i * 12 + 4 + g];
        float gi2 = gij[j * 12 + 8 + g] + gii[i * 12 + 8 + g];
        float rg  = sigm(gi0 + gh[g]);
        float z   = sigm(gi1 + gh[4 + g]);
        float nn2 = tanhf(gi2 + rg * gh[8 + g]);
        out[g] = ((1.f - z) * nn2 + z * e0v[g]) * aij;
    }
    float* ep = eout + eid * 4;   // base offset +24B: 8B-aligned -> two float2 stores
    *reinterpret_cast<float2*>(ep)     = make_float2(out[0], out[1]);
    *reinterpret_cast<float2*>(ep + 2) = make_float2(out[2], out[3]);
    wbuf[eid] = out[0];           // w = e[...,0], row-major for coalesced RNN reads
}

// ---------------- reservoir RNN ----------------
__global__ void k_rnn_init(const float* __restrict__ a0, float* __restrict__ abuf) {
    int j = blockIdx.x * 256 + threadIdx.x;
    abuf[j] = a0[j];
}

__global__ void k_rnn_step(const float* __restrict__ aprev, const float* __restrict__ obs,
                           const float* __restrict__ wbuf, float* __restrict__ anext,
                           float* __restrict__ aseq_out, int t) {
    int j = blockIdx.x * 256 + threadIdx.x;
    float av = (j < OBSD) ? obs[j] : aprev[j];
    aseq_out[j * 6 + t] = av;                     // a_seq column t = a_in at step t
    float acc = 0.f;
    for (int i = 0; i < NN; ++i) {                // w fits in L2 across all 5 steps
        float ain = (i < OBSD) ? obs[i] : aprev[i];
        acc += ain * wbuf[(size_t)i * NN + j];    // coalesced across j
    }
    anext[j] = tanhf(acc);
}

__global__ void k_fin(const float* __restrict__ afin, float* __restrict__ aseq_out,
                      float* __restrict__ action) {
    int j = blockIdx.x * 256 + threadIdx.x;
    aseq_out[j * 6 + 5] = afin[j];
    if (j >= NN - ACTD) {
        float v = afin[j];
        action[j - (NN - ACTD)] = (v != v) ? 0.f : v;  // NaN -> 0
    }
}

extern "C" void kernel_launch(void* const* d_in, const int* in_sizes, int n_in,
                              void* d_out, int out_size, void* d_ws, size_t ws_size,
                              hipStream_t stream) {
    (void)in_sizes; (void)n_in; (void)out_size; (void)ws_size;
    const float* obs   = (const float*)d_in[0];
    const float* A     = (const float*)d_in[1];
    // d_in[2] = maskA (only affects dead A_new path -> unused)
    const float* e0    = (const float*)d_in[3];
    const float* h0    = (const float*)d_in[4];
    const float* a0    = (const float*)d_in[5];
    const float* aseq0 = (const float*)d_in[6];
    const float* r     = (const float*)d_in[7];
    const float* Wpre_w = (const float*)d_in[8];
    const float* Wpre_b = (const float*)d_in[9];
    const float* Wq    = (const float*)d_in[10];
    const float* Wk    = (const float*)d_in[11];
    const float* Wv    = (const float*)d_in[12];
    const float* We    = (const float*)d_in[13];
    const float* Wo_w  = (const float*)d_in[14];
    const float* Wo_b  = (const float*)d_in[15];
    const float* nwih  = (const float*)d_in[16];
    const float* nwhh  = (const float*)d_in[17];
    const float* nbih  = (const float*)d_in[18];
    const float* nbhh  = (const float*)d_in[19];
    const float* ewih  = (const float*)d_in[20];
    const float* ewhh  = (const float*)d_in[21];
    const float* ebih  = (const float*)d_in[22];
    const float* ebhh  = (const float*)d_in[23];
    // d_in[24..31]: prune/add MLPs -> dead code w.r.t. outputs, skipped.

    // workspace layout (floats)
    float* wsf   = (float*)d_ws;
    float* isum  = wsf;              // N
    float* osum  = wsf + NN;         // N
    float* qbuf  = wsf + 2  * NN;    // 12N
    float* kbuf  = wsf + 14 * NN;    // 12N
    float* xt    = wsf + 26 * NN;    // 24N
    float* gij   = wsf + 50 * NN;    // 12N
    float* gii   = wsf + 62 * NN;    // 12N
    float* abuf0 = wsf + 74 * NN;    // N
    float* abuf1 = wsf + 75 * NN;    // N
    f16*   vbuf  = (f16*)(wsf + 76 * NN);       // B-frag image: 3*48*32*16 f16 = 24N floats
    float* wbuf  = wsf + 100 * NN;              // N*N floats (~9.4 MB)

    float* out     = (float*)d_out;
    float* action  = out;                           // 6
    float* eout    = out + 6;                       // N*N*4
    float* aseqout = out + 6 + (size_t)NN * NN * 4; // N*6

    k_rowsum<<<NN, 256, 0, stream>>>(A, isum);
    k_colsum<<<NN / 256, 256, 0, stream>>>(A, osum);
    k_node_prep<<<NN / 256, 256, 0, stream>>>(isum, osum, aseq0, h0, Wpre_w, Wpre_b,
                                              Wq, Wk, Wv, qbuf, kbuf, vbuf);
    k_attn<<<dim3(NN / 16, 3), 256, 0, stream>>>(qbuf, kbuf, vbuf, e0, A, We, xt);
    k_node_gru<<<NN / 256, 256, 0, stream>>>(xt, h0, Wo_w, Wo_b, nwih, nwhh, nbih, nbhh,
                                             ewih, ebih, aseq0, r, gij, gii);
    k_edge<<<(NN * (NN / 256)), 256, 0, stream>>>(e0, A, gij, gii, ewhh, ebhh, eout, wbuf);
    k_rnn_init<<<NN / 256, 256, 0, stream>>>(a0, abuf0);
    float* prev = abuf0; float* next = abuf1;
    for (int t = 0; t < 5; ++t) {
        k_rnn_step<<<NN / 256, 256, 0, stream>>>(prev, obs, wbuf, next, aseqout, t);
        float* tmp = prev; prev = next; next = tmp;
    }
    k_fin<<<NN / 256, 256, 0, stream>>>(prev, aseqout, action);
}
